// QuantumLSTM_65481071406812
// MI455X (gfx1250) — matrix operand order repaired and checked by reading, not compiled, not run on hardware
//
#include <hip/hip_runtime.h>

// ---------- types for WMMA fragments ----------
typedef __attribute__((ext_vector_type(16))) __bf16 v16bf;
typedef __attribute__((ext_vector_type(8)))  __bf16 v8bf;
typedef __attribute__((ext_vector_type(8)))  float  v8f;
typedef __attribute__((ext_vector_type(4)))  float  v4f;

#define TT 512
#define BB 256
#define DD 256
#define HH 256
#define KK 512   // D + H

// ---------------------------------------------------------------------------
// Prep: convert W (fp32 [4,H,512] row-major = [N=1024, K=512]) to bf16,
// zero cell state c and the t=0 hidden-state buffer. Runs every call so the
// scan is deterministic across graph replays.
// ---------------------------------------------------------------------------
__global__ __launch_bounds__(256) void qlstm_prep(const float* __restrict__ W,
                                                  unsigned short* __restrict__ Wb_us,
                                                  float* __restrict__ c,
                                                  unsigned short* __restrict__ h0_us) {
    __bf16* Wb = reinterpret_cast<__bf16*>(Wb_us);
    __bf16* h0 = reinterpret_cast<__bf16*>(h0_us);
    int i = blockIdx.x * blockDim.x + threadIdx.x;
    if (i < 4 * HH * KK) Wb[i] = (__bf16)W[i];
    if (i < BB * HH) { c[i] = 0.0f; h0[i] = (__bf16)0.0f; }
}

// A fragment for k-chunk: row m, elements j<8 -> K=kBase+8*lh+j,
// j>=8 -> K=kBase+16+8*lh+(j-8).  x part is fp32->bf16 (non-temporal:
// x is read-once streaming data, keep it out of the hot L2 set),
// h part is bf16 (hot, regular temporal).
__device__ __forceinline__ v16bf load_a(const float* __restrict__ xrow,
                                        const __bf16* __restrict__ hrow,
                                        int kBase, int lh) {
    v16bf a;
    if (kBase < DD) {
        const float* p0 = xrow + kBase + 8 * lh;
        const float* p1 = xrow + kBase + 16 + 8 * lh;
        v4f f0 = __builtin_nontemporal_load((const v4f*)p0);
        v4f f1 = __builtin_nontemporal_load((const v4f*)(p0 + 4));
        v4f f2 = __builtin_nontemporal_load((const v4f*)p1);
        v4f f3 = __builtin_nontemporal_load((const v4f*)(p1 + 4));
        #pragma unroll
        for (int j = 0; j < 4; ++j) {
            a[j]      = (__bf16)f0[j];
            a[4 + j]  = (__bf16)f1[j];
            a[8 + j]  = (__bf16)f2[j];
            a[12 + j] = (__bf16)f3[j];
        }
    } else {
        const int kk = kBase - DD;
        v8bf h0v = *(const v8bf*)(hrow + kk + 8 * lh);
        v8bf h1v = *(const v8bf*)(hrow + kk + 16 + 8 * lh);
        #pragma unroll
        for (int j = 0; j < 8; ++j) { a[j] = h0v[j]; a[8 + j] = h1v[j]; }
    }
    return a;
}

// B fragment for gate g, k-chunk: lane column n = g*256+qBase+ln,
// elements j<8 -> K = kBase + 8*lh + j, j>=8 -> K = kBase + 16 + 8*lh + (j-8).
// Weights are the hot recurrent set: regular temporal loads.
__device__ __forceinline__ v16bf load_b(const __bf16* __restrict__ wbase,
                                        int g, int kBase) {
    const __bf16* wp = wbase + (size_t)g * HH * KK + kBase;
    v8bf b0 = *(const v8bf*)wp;
    v8bf b1 = *(const v8bf*)(wp + 16);
    v16bf bm;
    #pragma unroll
    for (int j = 0; j < 8; ++j) { bm[j] = b0[j]; bm[8 + j] = b1[j]; }
    return bm;
}

// ---------------------------------------------------------------------------
// One LSTM step. Wave w owns batch tile mBase..+15 and q tile qBase..+15 for
// ALL FOUR gates; K-loop is software-pipelined: next chunk's A + 4 B
// fragments are kept in flight while the current 4 WMMAs execute.
// amdgpu_waves_per_eu(1,4): only ~2 waves/SIMD are resident anyway, so give
// the scheduler registers to hold the double buffer instead of collapsing it.
// Grid: 32 blocks x 256 threads = 256 waves = 16 m-tiles x 16 q-tiles.
// ---------------------------------------------------------------------------
__global__ __launch_bounds__(256)
__attribute__((amdgpu_waves_per_eu(1, 4)))
void qlstm_step(const float* __restrict__ x,
                const unsigned short* __restrict__ Wb_us,
                const float* __restrict__ bias,
                const unsigned short* __restrict__ hin_us,
                unsigned short* __restrict__ hout_us,
                float* __restrict__ c,
                float* __restrict__ out,
                int t, int last) {
    const __bf16* Wb    = reinterpret_cast<const __bf16*>(Wb_us);
    const __bf16* h_in  = reinterpret_cast<const __bf16*>(hin_us);
    __bf16*       h_out = reinterpret_cast<__bf16*>(hout_us);

    const int lane = threadIdx.x & 31;          // wave32
    const int wv   = threadIdx.x >> 5;
    const int w    = blockIdx.x * 8 + wv;       // 0..255
    const int mBase = (w >> 4) << 4;            // batch tile
    const int qBase = (w & 15) << 4;            // hidden-unit tile
    const int lh = lane >> 4;                   // lane half (A/B K split)
    const int ln = lane & 15;                   // A row / B column within tile

    v8f acc[4] = {v8f{}, v8f{}, v8f{}, v8f{}};

    const int m = mBase + ln;                                   // A-matrix row
    const float*  xrow  = x + ((size_t)t * BB + m) * DD;
    const __bf16* hrow  = h_in + (size_t)m * HH;
    const __bf16* wbase = Wb + (size_t)(qBase + ln) * KK + 8 * lh;

    // ---- software-pipelined K loop: 16 chunks of K=32 -------------------
    v16bf a_cur = load_a(xrow, hrow, 0, lh);
    v16bf b_cur[4];
    #pragma unroll
    for (int g = 0; g < 4; ++g) b_cur[g] = load_b(wbase, g, 0);

    #pragma unroll
    for (int kc = 0; kc < KK / 32; ++kc) {
        const int kNext = (kc + 1) * 32;
        v16bf a_nxt = a_cur;
        v16bf b_nxt[4] = {b_cur[0], b_cur[1], b_cur[2], b_cur[3]};
        if (kc + 1 < KK / 32) {
            a_nxt = load_a(xrow, hrow, kNext, lh);
            #pragma unroll
            for (int g = 0; g < 4; ++g) b_nxt[g] = load_b(wbase, g, kNext);
        }
        #pragma unroll
        for (int g = 0; g < 4; ++g) {
            acc[g] = __builtin_amdgcn_wmma_f32_16x16x32_bf16(
                false, a_cur, false, b_cur[g], (short)0, acc[g], false, false);
        }
        a_cur = a_nxt;
        #pragma unroll
        for (int g = 0; g < 4; ++g) b_cur[g] = b_nxt[g];
    }

    // ---- fused gate epilogue: C/D layout -> element r is (M = r + 8*lh, N = ln)
    const int q = qBase + ln;
    const float bf = bias[q];
    const float bi = bias[HH + q];
    const float bg = bias[2 * HH + q];
    const float bo = bias[3 * HH + q];

    #pragma unroll
    for (int r = 0; r < 8; ++r) {
        const int brow = mBase + r + 8 * lh;
        const int idx  = brow * HH + q;
        const float zf = acc[0][r] + bf;
        const float zi = acc[1][r] + bi;
        const float zg = acc[2][r] + bg;
        const float zo = acc[3][r] + bo;
        const float fg = 1.0f / (1.0f + __expf(-__cosf(zf)));
        const float ig = 1.0f / (1.0f + __expf(-__cosf(zi)));
        const float gg = tanhf(__cosf(zg));
        const float og = 1.0f / (1.0f + __expf(-__cosf(zo)));
        const float cn = fg * c[idx] + ig * gg;
        const float hn = og * tanhf(cn);
        c[idx] = cn;
        // outputs are write-once streaming: non-temporal so they don't evict
        // the hot recurrent set (Wb, h, c) from L2.
        __builtin_nontemporal_store(hn, &out[((size_t)t * BB + brow) * HH + q]);
        h_out[idx] = (__bf16)hn;
        if (last) {
            out[(size_t)TT * BB * HH + idx]           = hn;   // h_last
            out[(size_t)TT * BB * HH + BB * HH + idx] = cn;   // c_last
        }
    }
}

// ---------------------------------------------------------------------------
extern "C" void kernel_launch(void* const* d_in, const int* in_sizes, int n_in,
                              void* d_out, int out_size, void* d_ws, size_t ws_size,
                              hipStream_t stream) {
    const float* x    = (const float*)d_in[0];   // [T,B,D]
    const float* W    = (const float*)d_in[1];   // [4,H,D+H]
    const float* bias = (const float*)d_in[2];   // [4,H]
    float* out = (float*)d_out;                  // outputs | h_last | c_last

    char* ws = (char*)d_ws;
    unsigned short* Wb = (unsigned short*)ws;                           // 1 MB bf16
    float*          c  = (float*)(ws + 1048576);                        // 256 KB
    unsigned short* hA = (unsigned short*)(ws + 1048576 + 262144);      // 128 KB bf16
    unsigned short* hB = (unsigned short*)(ws + 1048576 + 262144 + 131072);

    qlstm_prep<<<(4 * HH * KK + 255) / 256, 256, 0, stream>>>(W, Wb, c, hA);

    for (int t = 0; t < TT; ++t) {
        unsigned short* hin  = (t & 1) ? hB : hA;
        unsigned short* hout = (t & 1) ? hA : hB;
        qlstm_step<<<32, 256, 0, stream>>>(x, Wb, bias, hin, hout, c, out,
                                           t, (t == TT - 1) ? 1 : 0);
    }
}
